// MultiHeadAttention_61787399520854
// MI455X (gfx1250) — matrix operand order repaired and checked by reading, not compiled, not run on hardware
//
#include <hip/hip_runtime.h>
#include <hip/hip_bf16.h>

// ---------------------------------------------------------------------------
// MHA forward for MI455X (gfx1250): bf16 WMMA everywhere, fp32 accumulation.
// Pipeline: cvt(fp32->bf16) -> QKV proj GEMM (+fused RoPE, V transposed)
//           -> flash attention (async K/V staging to LDS, online softmax,
//              causal) -> output proj GEMM.
// ---------------------------------------------------------------------------

typedef __attribute__((ext_vector_type(16))) __bf16 v16bf;
typedef __attribute__((ext_vector_type(8)))  __bf16 v8bf;
typedef __attribute__((ext_vector_type(4)))  __bf16 v4bf;
typedef __attribute__((ext_vector_type(8)))  float  v8f;

constexpr int Bsz = 2;
constexpr int S   = 2048;
constexpr int D   = 2048;
constexpr int H   = 16;
constexpr int DH  = 128;          // head dim
constexpr int BS  = Bsz * S;      // 4096 tokens total

// -------------------------- WMMA helper ------------------------------------
__device__ __forceinline__ v8f wmma_bf16(v16bf a, v16bf b, v8f c) {
  // D = A(16x32 bf16) * B(32x16 bf16) + C(16x16 f32)
  return __builtin_amdgcn_wmma_f32_16x16x32_bf16(
      /*neg_a=*/false, a, /*neg_b=*/false, b,
      /*c_mod=*/(short)0, c, /*reuse_a=*/false, /*reuse_b=*/false);
}

// A fragment (16x32, MxK): lane row = row0 + (lane&15);
// lane half 0 holds K {k0..k0+15}, half 1 holds {k0+16..k0+31}.
__device__ __forceinline__ v16bf load_a(const __bf16* __restrict__ A, int lda,
                                        int row0, int k0, int lo, int hi) {
  const __bf16* p = A + (size_t)(row0 + lo) * lda + k0 + hi * 16;
  v8bf x = *(const v8bf*)(p);        // 16B
  v8bf y = *(const v8bf*)(p + 8);    // 16B
  v16bf r;
#pragma unroll
  for (int i = 0; i < 8; ++i) { r[i] = x[i]; r[i + 8] = y[i]; }
  return r;
}

// B fragment (32x16, KxN) from a row-major [N,K] matrix (TN GEMM):
// lane column n = n0 + (lane&15); half picks K range of 16 contiguous elems.
__device__ __forceinline__ v16bf load_b(const __bf16* __restrict__ W, int ldw,
                                        int n0, int k0, int lo, int hi) {
  const __bf16* p = W + (size_t)(n0 + lo) * ldw + k0 + hi * 16;
  return *(const v16bf*)p;           // 32B contiguous
}

// ----------------------- fp32 -> bf16 conversion ---------------------------
__global__ void cvt_f32_bf16(const float* __restrict__ src,
                             __bf16* __restrict__ dst, int n4) {
  int i = blockIdx.x * blockDim.x + threadIdx.x;
  if (i < n4) {
    float4 v = *(const float4*)(src + 4 * i);
    v4bf o;
    o[0] = (__bf16)v.x; o[1] = (__bf16)v.y;
    o[2] = (__bf16)v.z; o[3] = (__bf16)v.w;
    *(v4bf*)(dst + 4 * i) = o;
  }
}

// ----------------------- QKV projection + RoPE -----------------------------
// grid = (BS/128, D/128, 3); block = 256 (8 waves, 4(M) x 2(N)).
// which==0 -> Q [b,h,s,dh] (+RoPE), which==1 -> K (+RoPE), which==2 -> V^T [b,h,dh,s].
__global__ void proj_qkv_kernel(const __bf16* __restrict__ X,
                                const __bf16* __restrict__ Wq,
                                const __bf16* __restrict__ Wk,
                                const __bf16* __restrict__ Wv,
                                const int*    __restrict__ tok,
                                __bf16* __restrict__ Qo,
                                __bf16* __restrict__ Ko,
                                __bf16* __restrict__ Vt) {
  const int which = blockIdx.z;
  const __bf16* W = (which == 0) ? Wq : (which == 1) ? Wk : Wv;

  const int tid  = threadIdx.x;
  const int lane = tid & 31, wid = tid >> 5;
  const int lo = lane & 15, hi = lane >> 4;
  const int wm = blockIdx.x * 128 + (wid >> 1) * 32;   // wave M base (32 rows)
  const int wn = blockIdx.y * 128 + (wid & 1) * 64;    // wave N base (64 cols)

  v8f acc[2][4];
#pragma unroll
  for (int a = 0; a < 2; ++a)
#pragma unroll
    for (int b = 0; b < 4; ++b) acc[a][b] = v8f{};

  for (int k0 = 0; k0 < D; k0 += 32) {
    v16bf a0 = load_a(X, D, wm,      k0, lo, hi);
    v16bf a1 = load_a(X, D, wm + 16, k0, lo, hi);
#pragma unroll
    for (int nt = 0; nt < 4; ++nt) {
      v16bf bf = load_b(W, D, wn + nt * 16, k0, lo, hi);
      acc[0][nt] = wmma_bf16(a0, bf, acc[0][nt]);
      acc[1][nt] = wmma_bf16(a1, bf, acc[1][nt]);
    }
  }

  // Epilogue: C layout row m = r + 8*hi, col n = lo within each 16x16 tile.
  if (which == 2) {
#pragma unroll
    for (int nt = 0; nt < 4; ++nt) {
      const int n  = wn + nt * 16 + lo;
      const int h  = n >> 7;
      const int dh = n & (DH - 1);
#pragma unroll
      for (int mt = 0; mt < 2; ++mt)
#pragma unroll
        for (int r = 0; r < 8; ++r) {
          const int m = wm + mt * 16 + r + 8 * hi;
          const int b = m >> 11;
          const int s = m & (S - 1);
          // V transposed: [b,h,dh,s]
          Vt[((size_t)(b * H + h) * DH + dh) * S + s] = (__bf16)acc[mt][nt][r];
        }
    }
  } else {
    __bf16* dst = (which == 0) ? Qo : Ko;          // [b,h,s,dh]
    // inv_freq = 10000^(-2i/128), i = dh/2 (uniform across rows)
    float invf[4];
#pragma unroll
    for (int nt = 0; nt < 4; ++nt) {
      const int dh = (wn + nt * 16 + lo) & (DH - 1);
      invf[nt] = __powf(10000.0f, -(float)(2 * (dh >> 1)) / (float)DH);
    }
#pragma unroll
    for (int mt = 0; mt < 2; ++mt)
#pragma unroll
      for (int r = 0; r < 8; ++r) {
        const int m = wm + mt * 16 + r + 8 * hi;   // global token row
        const int b = m >> 11;
        const int s = m & (S - 1);
        const float pos = (float)tok[m];
#pragma unroll
        for (int nt = 0; nt < 4; ++nt) {
          const int n  = wn + nt * 16 + lo;
          const int h  = n >> 7;
          const int dh = n & (DH - 1);
          const float x = acc[mt][nt][r];
          const float ang = pos * invf[nt];
          const float c = __cosf(ang), sn = __sinf(ang);
          const float p = __shfl_xor(x, 1, 32);    // partner feature (n^1)
          const float o = (dh & 1) ? (p * sn + x * c)    // odd:  x1*sin + x2*cos
                                   : (x * c - p * sn);   // even: x1*cos - x2*sin
          dst[((size_t)(b * H + h) * S + s) * DH + dh] = (__bf16)o;
        }
      }
  }
}

// ----------------------- flash attention (causal) --------------------------
// grid = (S/128, B*H); block = 256 (8 waves, each wave: 16 query rows).
// K/V tiles for each 32-key block are staged into LDS with async copies
// (double buffered, ASYNCcnt-tracked) and shared by all 8 waves.
__global__ void attn_kernel(const __bf16* __restrict__ Q,
                            const __bf16* __restrict__ K,
                            const __bf16* __restrict__ Vt,
                            __bf16* __restrict__ Ab) {
  const int qblk = blockIdx.x;
  const int bh   = blockIdx.y;
  const int b    = bh >> 4;
  const int h    = bh & (H - 1);
  const __bf16* Qbh = Q  + (size_t)bh * S * DH;
  const __bf16* Kbh = K  + (size_t)bh * S * DH;
  const __bf16* Vbh = Vt + (size_t)bh * DH * S;   // [dh, s]

  const int tid = threadIdx.x;
  const int lane = tid & 31, wid = tid >> 5;
  const int lo = lane & 15, hi = lane >> 4;
  const int q0 = qblk * 128 + wid * 16;           // this wave's 16 query rows

  // LDS: double-buffered K (32x128) / V^T (128x32) tiles, rows padded to
  // de-conflict banks (K row = 272B = 68 dwords, V row = 80B = 20 dwords),
  // plus per-wave P staging for the C->A fragment relayout.
  __shared__ __align__(16) __bf16 Kbuf[2][32][136];
  __shared__ __align__(16) __bf16 Vbuf[2][128][40];
  __shared__ float pl[8][16][36];
  float (*P)[36] = pl[wid];

  const unsigned kls = (unsigned)(uintptr_t)&Kbuf[0][0][0];
  const unsigned vls = (unsigned)(uintptr_t)&Vbuf[0][0][0];
  // copy assignment: 16KB / 256 threads = 64B per thread (2x32B chunks)
  const int krow = tid >> 3, kcol = (tid & 7) * 16;   // K: 32 rows x 256B
  const int vrow = tid >> 1, vcol = (tid & 1) * 16;   // V: 128 rows x 64B

  // Q fragments: 16x128 = 4 chunks of K=32, kept in registers.
  v16bf qa[4];
#pragma unroll
  for (int c = 0; c < 4; ++c) qa[c] = load_a(Qbh, DH, q0, c * 32, lo, hi);

  v8f acc[8];                                     // O: 16x128 fp32
#pragma unroll
  for (int j = 0; j < 8; ++j) acc[j] = v8f{};
  float mr[8], lr[8];
#pragma unroll
  for (int r = 0; r < 8; ++r) { mr[r] = -3.0e38f; lr[r] = 0.0f; }

  const float scl = 0.08838834764831845f;         // 1/sqrt(128)
  const int nkb = qblk * 4 + 4;                   // key blocks of 32 (causal bound)

  // async prefetch of key-block kb into LDS buffer kb&1 (4 ops per thread)
  auto prefetch = [&](int kb) {
    const int key0 = kb * 32;
    const unsigned buf = (unsigned)(kb & 1);
    const unsigned long long gk =
        (unsigned long long)(uintptr_t)(Kbh + (size_t)(key0 + krow) * DH + kcol);
    const unsigned lk = kls + (buf * 32u + (unsigned)krow) * 272u + (unsigned)kcol * 2u;
    asm volatile("global_load_async_to_lds_b128 %0, %1, off"
                 :: "v"(lk), "v"(gk) : "memory");
    asm volatile("global_load_async_to_lds_b128 %0, %1, off"
                 :: "v"(lk + 16u), "v"(gk + 16ull) : "memory");
    const unsigned long long gv =
        (unsigned long long)(uintptr_t)(Vbh + (size_t)vrow * S + key0 + vcol);
    const unsigned lv = vls + (buf * 128u + (unsigned)vrow) * 80u + (unsigned)vcol * 2u;
    asm volatile("global_load_async_to_lds_b128 %0, %1, off"
                 :: "v"(lv), "v"(gv) : "memory");
    asm volatile("global_load_async_to_lds_b128 %0, %1, off"
                 :: "v"(lv + 16u), "v"(gv + 16ull) : "memory");
  };

  prefetch(0);
  for (int kb = 0; kb < nkb; ++kb) {
    const int key0 = kb * 32;
    const int buf  = kb & 1;
    if (kb + 1 < nkb) {
      prefetch(kb + 1);                               // 4 new ops in flight
      asm volatile("s_wait_asynccnt 0x4" ::: "memory"); // current block resident
    } else {
      asm volatile("s_wait_asynccnt 0x0" ::: "memory");
    }
    __syncthreads();                                  // all waves' copies visible

    // ---- scores: two 16x16 tiles over 32 keys, contraction over dh=128 ----
    v8f s0 = v8f{}, s1 = v8f{};
#pragma unroll
    for (int c = 0; c < 4; ++c) {
      v16bf b0 = *(const v16bf*)&Kbuf[buf][lo][c * 32 + hi * 16];
      v16bf b1 = *(const v16bf*)&Kbuf[buf][16 + lo][c * 32 + hi * 16];
      s0 = wmma_bf16(qa[c], b0, s0);
      s1 = wmma_bf16(qa[c], b1, s1);
    }
    // ---- online softmax update (row = r + 8*hi, key col = lo) ----
#pragma unroll
    for (int r = 0; r < 8; ++r) {
      const int qrow = q0 + r + 8 * hi;
      float v0 = s0[r] * scl, v1 = s1[r] * scl;
      if (key0 + lo      > qrow) v0 = -3.0e38f;   // causal mask
      if (key0 + 16 + lo > qrow) v1 = -3.0e38f;
      float mx = fmaxf(v0, v1);
      mx = fmaxf(mx, __shfl_xor(mx, 1, 32));
      mx = fmaxf(mx, __shfl_xor(mx, 2, 32));
      mx = fmaxf(mx, __shfl_xor(mx, 4, 32));
      mx = fmaxf(mx, __shfl_xor(mx, 8, 32));
      const float mnew = fmaxf(mr[r], mx);
      const float corr = __expf(mr[r] - mnew);
      const float p0 = __expf(v0 - mnew);
      const float p1 = __expf(v1 - mnew);
      float rs = p0 + p1;
      rs += __shfl_xor(rs, 1, 32);
      rs += __shfl_xor(rs, 2, 32);
      rs += __shfl_xor(rs, 4, 32);
      rs += __shfl_xor(rs, 8, 32);
      lr[r] = lr[r] * corr + rs;
      mr[r] = mnew;
#pragma unroll
      for (int j = 0; j < 8; ++j) acc[j][r] *= corr;
      P[r + 8 * hi][lo]      = p0;                // stage P in C layout
      P[r + 8 * hi][16 + lo] = p1;
    }
    __builtin_amdgcn_wave_barrier();              // same-wave LDS ops are in-order
    // ---- re-layout P (16x32) into an A fragment ----
    v16bf pa;
#pragma unroll
    for (int j = 0; j < 8; ++j) {
      pa[j]     = (__bf16)P[lo][hi * 16 + j];     // K {0..7}/{16..23}
      pa[j + 8] = (__bf16)P[lo][hi * 16 + 8 + j]; // K {8..15}/{24..31}
    }
    __builtin_amdgcn_wave_barrier();
    // ---- O += P(16x32) * V(32x128) ----
#pragma unroll
    for (int j = 0; j < 8; ++j) {
      v16bf vb = *(const v16bf*)&Vbuf[buf][j * 16 + lo][hi * 16];
      acc[j] = wmma_bf16(pa, vb, acc[j]);
    }
    __syncthreads();                              // buffer reuse fence
  }

  // ---- normalize and write attention output in [b, s, h*128+dh] layout ----
#pragma unroll
  for (int r = 0; r < 8; ++r) {
    const int s = q0 + r + 8 * hi;
    const float rinv = 1.0f / lr[r];
#pragma unroll
    for (int j = 0; j < 8; ++j) {
      const int col = h * DH + j * 16 + lo;
      Ab[(size_t)(b * S + s) * D + col] = (__bf16)(acc[j][r] * rinv);
    }
  }
}

// ----------------------- output projection ---------------------------------
// out[m,n] = sum_v Ab[m,v] * Wo[n,v]; fp32 result straight to d_out.
__global__ void out_proj_kernel(const __bf16* __restrict__ Ab,
                                const __bf16* __restrict__ Wo,
                                float* __restrict__ out) {
  const int tid = threadIdx.x;
  const int lane = tid & 31, wid = tid >> 5;
  const int lo = lane & 15, hi = lane >> 4;
  const int wm = blockIdx.x * 128 + (wid >> 1) * 32;
  const int wn = blockIdx.y * 128 + (wid & 1) * 64;

  v8f acc[2][4];
#pragma unroll
  for (int a = 0; a < 2; ++a)
#pragma unroll
    for (int b = 0; b < 4; ++b) acc[a][b] = v8f{};

  for (int k0 = 0; k0 < D; k0 += 32) {
    v16bf a0 = load_a(Ab, D, wm,      k0, lo, hi);
    v16bf a1 = load_a(Ab, D, wm + 16, k0, lo, hi);
#pragma unroll
    for (int nt = 0; nt < 4; ++nt) {
      v16bf bf = load_b(Wo, D, wn + nt * 16, k0, lo, hi);
      acc[0][nt] = wmma_bf16(a0, bf, acc[0][nt]);
      acc[1][nt] = wmma_bf16(a1, bf, acc[1][nt]);
    }
  }
#pragma unroll
  for (int mt = 0; mt < 2; ++mt)
#pragma unroll
    for (int nt = 0; nt < 4; ++nt)
#pragma unroll
      for (int r = 0; r < 8; ++r) {
        const int m = wm + mt * 16 + r + 8 * hi;
        const int n = wn + nt * 16 + lo;
        out[(size_t)m * D + n] = acc[mt][nt][r];
      }
}

// ---------------------------------------------------------------------------
extern "C" void kernel_launch(void* const* d_in, const int* in_sizes, int n_in,
                              void* d_out, int out_size, void* d_ws, size_t ws_size,
                              hipStream_t stream) {
  (void)in_sizes; (void)n_in; (void)out_size; (void)ws_size;
  const float* X   = (const float*)d_in[0];
  const int*   tok = (const int*)d_in[1];
  // d_in[2]: mask (tril) — implemented as causal masking, not read.
  const float* Wq  = (const float*)d_in[3];
  const float* Wk  = (const float*)d_in[4];
  const float* Wv  = (const float*)d_in[5];
  const float* Wo  = (const float*)d_in[6];
  float* out = (float*)d_out;

  // Workspace layout (bf16), 112 MB total.
  char* ws = (char*)d_ws;
  const size_t szXD = (size_t)BS * D * sizeof(__bf16);  // 16 MB
  const size_t szW  = (size_t)D * D * sizeof(__bf16);   //  8 MB
  __bf16* Xb  = (__bf16*)(ws);
  __bf16* Wqb = (__bf16*)(ws + szXD);
  __bf16* Wkb = (__bf16*)(ws + szXD + szW);
  __bf16* Wvb = (__bf16*)(ws + szXD + 2 * szW);
  __bf16* Wob = (__bf16*)(ws + szXD + 3 * szW);
  __bf16* Qb  = (__bf16*)(ws + szXD + 4 * szW);
  __bf16* Kb  = (__bf16*)(ws + 2 * szXD + 4 * szW);
  __bf16* Vt  = (__bf16*)(ws + 3 * szXD + 4 * szW);
  __bf16* Ab  = (__bf16*)(ws + 4 * szXD + 4 * szW);

  // 1) fp32 -> bf16
  {
    const int nX4 = BS * D / 4, nW4 = D * D / 4;
    cvt_f32_bf16<<<(nX4 + 255) / 256, 256, 0, stream>>>(X,  Xb,  nX4);
    cvt_f32_bf16<<<(nW4 + 255) / 256, 256, 0, stream>>>(Wq, Wqb, nW4);
    cvt_f32_bf16<<<(nW4 + 255) / 256, 256, 0, stream>>>(Wk, Wkb, nW4);
    cvt_f32_bf16<<<(nW4 + 255) / 256, 256, 0, stream>>>(Wv, Wvb, nW4);
    cvt_f32_bf16<<<(nW4 + 255) / 256, 256, 0, stream>>>(Wo, Wob, nW4);
  }
  // 2) QKV projection (+RoPE, V transposed)
  proj_qkv_kernel<<<dim3(BS / 128, D / 128, 3), 256, 0, stream>>>(
      Xb, Wqb, Wkb, Wvb, tok, Qb, Kb, Vt);
  // 3) flash attention (async K/V LDS staging)
  attn_kernel<<<dim3(S / 128, Bsz * H), 256, 0, stream>>>(Qb, Kb, Vt, Ab);
  // 4) output projection
  out_proj_kernel<<<dim3(BS / 128, D / 128), 256, 0, stream>>>(Ab, Wob, out);
}
